// RVQuantizer_70007966924839
// MI455X (gfx1250) — matrix operand order repaired and checked by reading, not compiled, not run on hardware
//
#include <hip/hip_runtime.h>
#include <hip/hip_bf16.h>
#include <math.h>

typedef __attribute__((ext_vector_type(16))) __bf16 v16bf;
typedef __attribute__((ext_vector_type(8)))  float  v8f;

#define NPTS   (8 * 32768)   // B*N = 262144 points
#define DIM    32
#define NQ     10
#define NCODE  512
#define WG_PTS 128           // 8 waves * 16 points
#define TPB    256

// LDS partition (dynamic): bf16 hi codebook, bf16 lo codebook, f32 norms, per-row idx
#define LDS_HI_OFF   0
#define LDS_LO_OFF   (NCODE * DIM * 2)                 // 32768
#define LDS_NRM_OFF  (LDS_LO_OFF + NCODE * DIM * 2)    // 65536
#define LDS_IDX_OFF  (LDS_NRM_OFF + NCODE * 4)         // 67584
#define LDS_BYTES    (LDS_IDX_OFF + WG_PTS * 4)        // 68096

__global__ void __launch_bounds__(TPB)
rvq_kernel(const float* __restrict__ z, const float* __restrict__ codebooks,
           float* __restrict__ outq, int* __restrict__ idx_out,
           float* __restrict__ commit) {
    extern __shared__ char smem[];
    __bf16* cb_hi = reinterpret_cast<__bf16*>(smem + LDS_HI_OFF);
    __bf16* cb_lo = reinterpret_cast<__bf16*>(smem + LDS_LO_OFF);
    float*  norms = reinterpret_cast<float*>(smem + LDS_NRM_OFF);
    int*    idxl  = reinterpret_cast<int*>(smem + LDS_IDX_OFF);

    const int tid  = threadIdx.x;
    const int wave = tid >> 5;
    const int lane = tid & 31;
    const int n16  = lane & 15;      // row m (for residual) / col n (for B/D tiles)
    const int h    = lane >> 4;      // lane half

    const int wgBase  = blockIdx.x * WG_PTS + wave * 16;
    const int myPoint = wgBase + n16;

    if (blockIdx.x == 0 && tid < NQ) commit[tid] = 0.0f;

    // ---- load z row into registers in WMMA A layout ----
    // r[0..7]  = k = h*8 + {0..7};  r[8..15] = k = 16 + h*8 + {0..7}
    float r[16] __attribute__((aligned(16)));
    {
        const float* zp = z + (size_t)myPoint * DIM;
        const float4* c0 = reinterpret_cast<const float4*>(zp + h * 8);
        const float4* c1 = reinterpret_cast<const float4*>(zp + 16 + h * 8);
        *reinterpret_cast<float4*>(&r[0])  = c0[0];
        *reinterpret_cast<float4*>(&r[4])  = c0[1];
        *reinterpret_cast<float4*>(&r[8])  = c1[0];
        *reinterpret_cast<float4*>(&r[12]) = c1[1];
    }

    for (int s = 0; s < NQ; ++s) {
        const float* cbg = codebooks + (size_t)s * NCODE * DIM;

        // ---- cooperative: stage codebook -> LDS as bf16 hi/lo ----
        for (int e = tid; e < NCODE * DIM; e += TPB) {
            float f = cbg[e];
            __bf16 hi = (__bf16)f;
            cb_hi[e] = hi;
            cb_lo[e] = (__bf16)(f - (float)hi);
        }
        // ---- cooperative: exact fp32 code norms ----
        for (int c = tid; c < NCODE; c += TPB) {
            const float4* row = reinterpret_cast<const float4*>(cbg + c * DIM);
            float s2 = 0.0f;
#pragma unroll
            for (int q = 0; q < 8; ++q) {
                float4 v = row[q];
                s2 += v.x * v.x + v.y * v.y + v.z * v.z + v.w * v.w;
            }
            norms[c] = s2;
        }
        __syncthreads();

        // ---- split residual to bf16 hi/lo (A operand) ----
        v16bf a_hi, a_lo;
#pragma unroll
        for (int i = 0; i < 16; ++i) {
            __bf16 hi = (__bf16)r[i];
            a_hi[i] = hi;
            a_lo[i] = (__bf16)(r[i] - (float)hi);
        }

        float bestS[8];
        int   bestI[8];
#pragma unroll
        for (int j = 0; j < 8; ++j) { bestS[j] = INFINITY; bestI[j] = 0x7fffffff; }

        // ---- 32 code tiles, 2 at a time (two independent WMMA chains in flight) ----
        // score = ||c||^2 - 2*dot, dot via bf16x3 WMMA (fp32-faithful)
        for (int t = 0; t < 32; t += 2) {
            const int cA = t * 16 + n16;
            const int cB = cA + 16;
            // B layout: lane holds column n16 of the tile, K = h*8+{0..7}, then 16+h*8+{0..7}
            typedef __attribute__((ext_vector_type(8))) __bf16 v8bf;
#define LDB(p) __builtin_shufflevector(*reinterpret_cast<const v8bf*>((p) + h * 8), *reinterpret_cast<const v8bf*>((p) + 16 + h * 8), 0,1,2,3,4,5,6,7,8,9,10,11,12,13,14,15)
            const v16bf bA_hi = LDB(cb_hi + cA * DIM);
            const v16bf bA_lo = LDB(cb_lo + cA * DIM);
            const v16bf bB_hi = LDB(cb_hi + cB * DIM);
            const v16bf bB_lo = LDB(cb_lo + cB * DIM);
#undef LDB
            v8f accA = {};
            v8f accB = {};
            accA = __builtin_amdgcn_wmma_f32_16x16x32_bf16(false, a_hi, false, bA_hi,
                                                           (short)0, accA, false, false);
            accB = __builtin_amdgcn_wmma_f32_16x16x32_bf16(false, a_hi, false, bB_hi,
                                                           (short)0, accB, false, false);
            accA = __builtin_amdgcn_wmma_f32_16x16x32_bf16(false, a_hi, false, bA_lo,
                                                           (short)0, accA, false, false);
            accB = __builtin_amdgcn_wmma_f32_16x16x32_bf16(false, a_hi, false, bB_lo,
                                                           (short)0, accB, false, false);
            accA = __builtin_amdgcn_wmma_f32_16x16x32_bf16(false, a_lo, false, bA_hi,
                                                           (short)0, accA, false, false);
            accB = __builtin_amdgcn_wmma_f32_16x16x32_bf16(false, a_lo, false, bB_hi,
                                                           (short)0, accB, false, false);
            const float nrmA = norms[cA];
            const float nrmB = norms[cB];
#pragma unroll
            for (int j = 0; j < 8; ++j) {
                float scA = fmaf(-2.0f, accA[j], nrmA);
                if (scA < bestS[j]) { bestS[j] = scA; bestI[j] = cA; }  // strict <: first idx wins
            }
#pragma unroll
            for (int j = 0; j < 8; ++j) {
                float scB = fmaf(-2.0f, accB[j], nrmB);
                if (scB < bestS[j]) { bestS[j] = scB; bestI[j] = cB; }
            }
        }

        // ---- branch-free butterfly argmin across the 16 lanes of each half ----
#pragma unroll
        for (int m = 1; m <= 8; m <<= 1) {
#pragma unroll
            for (int j = 0; j < 8; ++j) {
                float os = __shfl_xor(bestS[j], m, 32);
                int   oi = __shfl_xor(bestI[j], m, 32);
                int lt  = (os < bestS[j]) ? 1 : 0;
                int eq  = (os == bestS[j]) ? 1 : 0;
                int ilt = (oi < bestI[j]) ? 1 : 0;
                int take = lt | (eq & ilt);
                bestS[j] = take ? os : bestS[j];
                bestI[j] = take ? oi : bestI[j];
            }
        }
        if (n16 == 0) {
#pragma unroll
            for (int j = 0; j < 8; ++j) idxl[wave * 16 + h * 8 + j] = bestI[j];
        }
        __syncthreads();

        const int myIdx = idxl[wave * 16 + n16];

        // ---- exact fp32 residual update via gather from global (L2-resident) ----
        {
            const float* crow = cbg + (size_t)myIdx * DIM;
            const float4* g0 = reinterpret_cast<const float4*>(crow + h * 8);
            const float4* g1 = reinterpret_cast<const float4*>(crow + 16 + h * 8);
            float4 q0 = g0[0], q1 = g0[1], q2 = g1[0], q3 = g1[1];
            r[0]  -= q0.x; r[1]  -= q0.y; r[2]  -= q0.z; r[3]  -= q0.w;
            r[4]  -= q1.x; r[5]  -= q1.y; r[6]  -= q1.z; r[7]  -= q1.w;
            r[8]  -= q2.x; r[9]  -= q2.y; r[10] -= q2.z; r[11] -= q2.w;
            r[12] -= q3.x; r[13] -= q3.y; r[14] -= q3.z; r[15] -= q3.w;
        }

        if (lane < 16) idx_out[(size_t)(wgBase + lane) * NQ + s] = myIdx;

        __syncthreads();  // protect LDS before next stage's fill
    }

    // ---- quantized = z - final residual ----
    {
        const float* zp = z + (size_t)myPoint * DIM;
        float* op = outq + (size_t)myPoint * DIM;
        const float4* c0 = reinterpret_cast<const float4*>(zp + h * 8);
        const float4* c1 = reinterpret_cast<const float4*>(zp + 16 + h * 8);
        float4 z0 = c0[0], z1 = c0[1], z2 = c1[0], z3 = c1[1];
        float4 o0 = make_float4(z0.x - r[0],  z0.y - r[1],  z0.z - r[2],  z0.w - r[3]);
        float4 o1 = make_float4(z1.x - r[4],  z1.y - r[5],  z1.z - r[6],  z1.w - r[7]);
        float4 o2 = make_float4(z2.x - r[8],  z2.y - r[9],  z2.z - r[10], z2.w - r[11]);
        float4 o3 = make_float4(z3.x - r[12], z3.y - r[13], z3.z - r[14], z3.w - r[15]);
        reinterpret_cast<float4*>(op + h * 8)[0]      = o0;
        reinterpret_cast<float4*>(op + h * 8)[1]      = o1;
        reinterpret_cast<float4*>(op + 16 + h * 8)[0] = o2;
        reinterpret_cast<float4*>(op + 16 + h * 8)[1] = o3;
    }
}

extern "C" void kernel_launch(void* const* d_in, const int* in_sizes, int n_in,
                              void* d_out, int out_size, void* d_ws, size_t ws_size,
                              hipStream_t stream) {
    const float* z         = (const float*)d_in[0];   // [8, 32768, 32] f32
    const float* codebooks = (const float*)d_in[1];   // [10, 512, 32] f32

    float* outF    = (float*)d_out;
    float* outq    = outF;                                    // [P, 32] f32
    int*   idx_out = (int*)(outF + (size_t)NPTS * DIM);       // [P, 10] i32 (bitwise)
    float* commit  = outF + (size_t)NPTS * DIM + (size_t)NPTS * NQ;  // [10] zeros

    const int grid = NPTS / WG_PTS;  // 2048
    rvq_kernel<<<grid, TPB, LDS_BYTES, stream>>>(z, codebooks, outq, idx_out, commit);
}
